// GCNModel_22170621182395
// MI455X (gfx1250) — matrix operand order repaired and checked by reading, not compiled, not run on hardware
//
#include <hip/hip_runtime.h>

#define HD   256
#define EPSF 1e-5f

typedef __attribute__((ext_vector_type(16))) __bf16       v16bf;
typedef __attribute__((ext_vector_type(8)))  float        v8f;
typedef __attribute__((ext_vector_type(8)))  int          v8i;
typedef __attribute__((ext_vector_type(4)))  int          i32x4;
typedef __attribute__((ext_vector_type(8)))  int          i32x8;
typedef __attribute__((ext_vector_type(4)))  unsigned int u32x4;

#if defined(__AMDGCN__) && __has_builtin(__builtin_amdgcn_tensor_load_to_lds) && \
    __has_builtin(__builtin_amdgcn_s_wait_tensorcnt)
#define USE_TDM 1
#else
#define USE_TDM 0
#endif

// Packed W size: 8 ksteps * 16 coltiles * 32 lanes * 16 bf16 = 64K bf16 = 128 KB
#define WPACK_BYTES (8 * 16 * 32 * 16 * 2)

struct bpair { int4 lo, hi; };
static __device__ __forceinline__ v16bf make_frag(int4 lo, int4 hi) {
    bpair p{lo, hi};
    return __builtin_bit_cast(v16bf, p);
}

// ---------------- utility kernels ----------------

__global__ void k_fill(float* __restrict__ p, float v, int n) {
    int i = blockIdx.x * blockDim.x + threadIdx.x;
    if (i < n) p[i] = v;
}

__global__ void k_deg(const int* __restrict__ dst, float* __restrict__ deg, int E) {
    int i = blockIdx.x * blockDim.x + threadIdx.x;
    if (i < E) atomicAdd(&deg[dst[i]], 1.0f);
}

__global__ void k_rsqrt_inplace(float* __restrict__ p, int n) {
    int i = blockIdx.x * blockDim.x + threadIdx.x;
    if (i < n) p[i] = rsqrtf(p[i]);
}

__global__ void k_f32_to_bf16(const float* __restrict__ in, __bf16* __restrict__ out, int n) {
    int i = blockIdx.x * blockDim.x + threadIdx.x;
    if (i < n) out[i] = (__bf16)in[i];
}

// Pack W[256,256] (f32 row-major, K x N) into WMMA B-fragment order:
// P[((t*16 + tn)*32 + lane)*16 + j] = bf16( W[(t*32 + (lane>>4)*16 + j)*256 + tn*16 + (lane&15)] )
__global__ void k_pack_w(const float* __restrict__ W, __bf16* __restrict__ P) {
    int id = blockIdx.x * blockDim.x + threadIdx.x;   // 4096 = 8*16*32
    if (id >= 8 * 16 * 32) return;
    int lane = id & 31;
    int tn   = (id >> 5) & 15;
    int t    = id >> 9;
    int half = lane >> 4, l15 = lane & 15;
    __bf16* o = P + (size_t)id * 16;
    #pragma unroll
    for (int j = 0; j < 16; ++j)
        o[j] = (__bf16)W[(size_t)(t * 32 + half * 16 + j) * HD + tn * 16 + l15];
}

// ---------------- WMMA GEMM: C[M,256] = A[M,256] @ W[256,256] ----------------
// One wave computes a 16x128 strip (8 col-tiles); fully unrolled -> 64 WMMAs.
// Packed W (128 KB) is TDM-copied to LDS once per block when available.

__global__ void k_gemm_wmma(const __bf16* __restrict__ Abf,
                            const __bf16* __restrict__ Wpack,
                            float* __restrict__ C, int M) {
    extern __shared__ char smem[];

#if USE_TDM
    if ((threadIdx.x >> 5) == 0) {
        unsigned long long ga = (unsigned long long)(const void*)Wpack;
        u32x4 g0 = { 1u,                                   // count=1 (valid descriptor)
                     0u,                                   // lds_addr = 0 (dynamic LDS base)
                     (unsigned)(ga & 0xffffffffull),       // global_addr[31:0]
                     (unsigned)((ga >> 32) & 0x01ffffffu) | 0x80000000u };  // addr[56:32] | type=2
        i32x8 g1 = { (int)0x00030000u,                     // data_size=3 (8B), mask=0, flags=0
                     (int)((WPACK_BYTES / 8) << 16),       // tensor_dim0[15:0] in bits 63:48
                     (int)(1u << 16),                      // tensor_dim0 hi=0 | tensor_dim1=1
                     (int)((WPACK_BYTES / 8) << 16),       // tensor_dim1 hi=0 | tile_dim0
                     0,                                    // tile_dim1=0 (1-D), tile_dim2=0
                     (int)(WPACK_BYTES / 8),               // tensor_dim0_stride[31:0]
                     0, 0 };
        i32x4 g2 = { 0, 0, 0, 0 };
        i32x4 g3 = { 0, 0, 0, 0 };
#if __clang_major__ >= 23
        i32x8 g4 = { 0, 0, 0, 0, 0, 0, 0, 0 };
        __builtin_amdgcn_tensor_load_to_lds(g0, g1, g2, g3, g4, 0);
#else
        __builtin_amdgcn_tensor_load_to_lds(g0, g1, g2, g3, 0);
#endif
        __builtin_amdgcn_s_wait_tensorcnt(0);
    }
    __syncthreads();
#define BFRAG(off) (*(const v8i*)(smem + (off)))
#else
#define BFRAG(off) (*(const v8i*)((const char*)Wpack + (off)))
#endif

    int lane  = threadIdx.x & 31;
    int wave  = (blockIdx.x * blockDim.x + threadIdx.x) >> 5;
    int strip = wave >> 1;          // 16-row strip index
    int nh    = wave & 1;           // column half: tiles nh*8 .. nh*8+7
    if (strip * 16 >= M) return;    // uniform per wave; after the barrier

    int half = lane >> 4, l15 = lane & 15;
    const __bf16* arow = Abf + (size_t)(strip * 16 + l15) * HD;

    v8f acc[8] = {};

    #pragma unroll
    for (int t = 0; t < 8; ++t) {
        int ka = t * 32 + half * 16;    // A frag: K runs [ka, ka+16)
        int4 alo = *(const int4*)(arow + ka);
        int4 ahi = *(const int4*)(arow + ka + 8);
        v16bf a = make_frag(alo, ahi);
        #pragma unroll
        for (int n = 0; n < 8; ++n) {
            int tn = nh * 8 + n;
            unsigned boff = (unsigned)(((t * 16 + tn) * 32 + lane) * 32);
            v16bf b = __builtin_bit_cast(v16bf, BFRAG(boff));
            acc[n] = __builtin_amdgcn_wmma_f32_16x16x32_bf16(
                         false, a, false, b, (short)0, acc[n], false, false);
        }
    }

    // C layout: VGPR r -> row = r + 8*half, col = l15
    #pragma unroll
    for (int n = 0; n < 8; ++n) {
        float* cb = C + (size_t)(strip * 16 + half * 8) * HD + (nh * 8 + n) * 16 + l15;
        #pragma unroll
        for (int r = 0; r < 8; ++r)
            cb[(size_t)r * HD] = acc[n][r];
    }
#undef BFRAG
}

// ---------------- aggregation ----------------

// agg[i,f] = bias[f] + h[i,f] * dinv[i]^2     (self-loop + bias init)
__global__ void k_init_agg(const float* __restrict__ h, const float* __restrict__ dinv,
                           const float* __restrict__ bias, float* __restrict__ agg, int M) {
    int idx = blockIdx.x * blockDim.x + threadIdx.x;
    if (idx < M * HD) {
        int i = idx >> 8, f = idx & 255;
        float di = dinv[i];
        agg[idx] = bias[f] + h[idx] * di * di;
    }
}

// one wave per edge; 32-lane coalesced feature chunks, f32 atomics (L2-resident)
__global__ void k_scatter(const int* __restrict__ src, const int* __restrict__ dst,
                          const float* __restrict__ dinv, const float* __restrict__ h,
                          float* __restrict__ agg, int E) {
    int wave = (blockIdx.x * blockDim.x + threadIdx.x) >> 5;
    int lane = threadIdx.x & 31;
    if (wave >= E) return;
    int s = src[wave], d = dst[wave];
    float w = dinv[s] * dinv[d];
    const float* hr = h + (size_t)s * HD;
    float* ar = agg + (size_t)d * HD;
    #pragma unroll
    for (int c = 0; c < 8; ++c) {
        int f = lane + 32 * c;
        atomicAdd(&ar[f], hr[f] * w);
    }
}

// ---------------- BatchNorm ----------------

__global__ void k_bn_stats(const float* __restrict__ a, float* __restrict__ sums, int M) {
    int f  = threadIdx.x;
    int r0 = blockIdx.x * 128;
    int r1 = min(r0 + 128, M);
    float s = 0.f, q = 0.f;
    for (int r = r0; r < r1; ++r) {
        float v = a[(size_t)r * HD + f];
        s += v; q += v * v;
    }
    atomicAdd(&sums[f], s);
    atomicAdd(&sums[HD + f], q);
}

__global__ void k_bn_final(const float* __restrict__ sums, const float* __restrict__ g,
                           const float* __restrict__ be, float* __restrict__ ss, int M) {
    int f = threadIdx.x;
    float inv = 1.0f / (float)M;
    float mu  = sums[f] * inv;
    float var = sums[HD + f] * inv - mu * mu;       // biased variance (torch BN)
    float sc  = g[f] * rsqrtf(var + EPSF);
    ss[f]      = sc;
    ss[HD + f] = be[f] - mu * sc;
}

// BN + ReLU, emitting bf16 (next layer's GEMM A operand / head input)
__global__ void k_bn_apply_relu_bf16(const float* __restrict__ a, const float* __restrict__ ss,
                                     __bf16* __restrict__ y, int M) {
    int idx = blockIdx.x * blockDim.x + threadIdx.x;
    if (idx < M * HD) {
        int f = idx & 255;
        float v = a[idx] * ss[f] + ss[HD + f];
        y[idx] = (__bf16)(v > 0.f ? v : 0.f);
    }
}

// ---------------- head: out[i] = h[i,:] . Wl + bl ----------------

__global__ void k_head(const __bf16* __restrict__ h, const float* __restrict__ Wl,
                       const float* __restrict__ bl, float* __restrict__ out, int M) {
    int wave = (blockIdx.x * blockDim.x + threadIdx.x) >> 5;
    int lane = threadIdx.x & 31;
    if (wave >= M) return;
    const __bf16* hr = h + (size_t)wave * HD;
    float s = 0.f;
    #pragma unroll
    for (int c = 0; c < 8; ++c) {
        int f = lane + 32 * c;
        s += (float)hr[f] * Wl[f];
    }
    #pragma unroll
    for (int off = 16; off > 0; off >>= 1)
        s += __shfl_xor(s, off, 32);
    if (lane == 0) out[wave] = s + bl[0];
}

// ---------------- launch ----------------

extern "C" void kernel_launch(void* const* d_in, const int* in_sizes, int n_in,
                              void* d_out, int out_size, void* d_ws, size_t ws_size,
                              hipStream_t stream) {
    const float* x   = (const float*)d_in[0];
    const int*   ei  = (const int*)d_in[1];
    const float* Wm[3] = { (const float*)d_in[2],  (const float*)d_in[6],  (const float*)d_in[10] };
    const float* bm[3] = { (const float*)d_in[3],  (const float*)d_in[7],  (const float*)d_in[11] };
    const float* gm[3] = { (const float*)d_in[4],  (const float*)d_in[8],  (const float*)d_in[12] };
    const float* em[3] = { (const float*)d_in[5],  (const float*)d_in[9],  (const float*)d_in[13] };
    const float* Wl  = (const float*)d_in[14];
    const float* bl  = (const float*)d_in[15];
    float* out = (float*)d_out;

    const int Nn = in_sizes[0] / HD;      // nodes
    const int E  = in_sizes[1] / 2;       // edges
    const int* src = ei;
    const int* dst = ei + E;

    // workspace layout (bytes, 256-aligned)
    char* ws = (char*)d_ws;
    size_t nhF32 = (size_t)Nn * HD * sizeof(float);
    size_t nhBF  = (size_t)Nn * HD * sizeof(__bf16);
    auto align256 = [](size_t v) { return (v + 255) & ~(size_t)255; };
    size_t off = 0;
    float*  B0    = (float*)(ws + off);  off += align256(nhF32);   // gemm out (f32)
    float*  B1    = (float*)(ws + off);  off += align256(nhF32);   // agg (f32)
    __bf16* B2bf  = (__bf16*)(ws + off); off += align256(nhBF);    // BN out (bf16)
    __bf16* Xbf   = (__bf16*)(ws + off); off += align256(nhBF);    // input x as bf16
    __bf16* Wpack = (__bf16*)(ws + off); off += align256(WPACK_BYTES);
    float*  dinv  = (float*)(ws + off);  off += align256((size_t)Nn * sizeof(float));
    float*  sums  = (float*)(ws + off);  off += align256(2 * HD * sizeof(float));
    float*  ss    = (float*)(ws + off);  off += align256(2 * HD * sizeof(float));
    (void)ws_size; (void)n_in; (void)out_size;

    const int T = 256;
    auto cdiv = [](int a, int b) { return (a + b - 1) / b; };

    // degree with self-loops -> dinv
    k_fill<<<cdiv(Nn, T), T, 0, stream>>>(dinv, 1.0f, Nn);
    k_deg<<<cdiv(E, T), T, 0, stream>>>(dst, dinv, E);
    k_rsqrt_inplace<<<cdiv(Nn, T), T, 0, stream>>>(dinv, Nn);

    // input activations -> bf16 once
    k_f32_to_bf16<<<cdiv(Nn * HD, T), T, 0, stream>>>(x, Xbf, Nn * HD);

    int tilesM   = cdiv(Nn, 16);
    int gemmBlks = cdiv(tilesM * 2, 8);         // wave = 16x128 strip; 8 waves/block
    int elemBlks = cdiv(Nn * HD, T);
    int edgeBlks = cdiv(E, 8);                  // wave per edge
    int statBlks = cdiv(Nn, 128);

    const __bf16* cur = Xbf;
    for (int l = 0; l < 3; ++l) {
        k_pack_w<<<cdiv(8 * 16 * 32, T), T, 0, stream>>>(Wm[l], Wpack);
        k_gemm_wmma<<<gemmBlks, T, WPACK_BYTES, stream>>>(cur, Wpack, B0, Nn);
        k_init_agg<<<elemBlks, T, 0, stream>>>(B0, dinv, bm[l], B1, Nn);
        k_scatter<<<edgeBlks, T, 0, stream>>>(src, dst, dinv, B0, B1, E);
        k_fill<<<2, T, 0, stream>>>(sums, 0.0f, 2 * HD);
        k_bn_stats<<<statBlks, T, 0, stream>>>(B1, sums, Nn);
        k_bn_final<<<1, T, 0, stream>>>(sums, gm[l], em[l], ss, Nn);
        k_bn_apply_relu_bf16<<<elemBlks, T, 0, stream>>>(B1, ss, B2bf, Nn);
        cur = B2bf;
    }

    k_head<<<cdiv(Nn, 8), T, 0, stream>>>(B2bf, Wl, bl, out, Nn);
}
